// TriangleModule_22574348108025
// MI455X (gfx1250) — matrix operand order repaired and checked by reading, not compiled, not run on hardware
//
#include <hip/hip_runtime.h>
#include <stdint.h>

// ---------------------------------------------------------------------------
// TriangleModule forward for MI455X (gfx1250, wave32, WMMA bf16->f32).
//
// Shapes: N1=N2=320, C=256, CT=16, H=4, CH=32, L=2, E=8000, M=N1*N2=102400.
//
// Input pointer order assumption:
//   d_in[0..7] = node1,node2,ef1,ef2,e1_src,e1_dst,e2_src,e2_dst (dict order)
//   d_in[8..]  = params pytree leaves in JAX order (dicts sorted by key):
//     attn[0]: bf,bg,ln_b,ln_g,wf,wg,wk,wq,wv    (9 leaves)  idx 8..16
//     attn[1]: same                                           idx 17..25
//     out:     b,w                                            idx 26..27
//     trans:   b1,b2,ln_b,ln_g,w1,w2                          idx 28..33
//     trig[0]: asb,asw,b1,b2,egb,egw,gb1,gb2,gw1,gw2,
//              ln_b,ln_g,lnc_b,lnc_g,w1,w2       (16 leaves)  idx 34..49
//     trig[1]: same                                           idx 50..65
// ---------------------------------------------------------------------------

#define NN1 320
#define NN2 320
#define CC  256
#define CT  16
#define HH  4
#define CHD 32
#define EE  8000
#define MROWS (NN1*NN2)   // 102400

// Try the gfx1250 async global->LDS path if the toolchain exposes it.
#ifndef TRY_ASYNC_LDS
#define TRY_ASYNC_LDS 1
#endif
#if TRY_ASYNC_LDS && defined(__has_builtin)
#if __has_builtin(__builtin_amdgcn_global_load_async_to_lds_b128) && \
    __has_builtin(__builtin_amdgcn_s_wait_asynccnt)
#define HAVE_ASYNC_LDS 1
#endif
#endif
#ifndef HAVE_ASYNC_LDS
#define HAVE_ASYNC_LDS 0
#endif

typedef __attribute__((ext_vector_type(16))) __bf16 v16bf;
typedef __attribute__((ext_vector_type(8)))  float  v8f;
typedef __attribute__((ext_vector_type(4)))  int    v4i;
#if HAVE_ASYNC_LDS
typedef __attribute__((address_space(1))) v4i gv4i;   // global int4
typedef __attribute__((address_space(3))) v4i lv4i;   // LDS int4
#endif

union FragU { v16bf v; uint32_t u[8]; uint16_t h[16]; };

__device__ __forceinline__ uint16_t f2bf(float x) {
  uint32_t u = __float_as_uint(x);
  uint32_t r = (u + 0x7FFFu + ((u >> 16) & 1u)) >> 16;
  return (uint16_t)r;
}
__device__ __forceinline__ float bf2f(uint16_t h) {
  return __uint_as_float(((uint32_t)h) << 16);
}
__device__ __forceinline__ float sigf(float x) { return 1.0f / (1.0f + __expf(-x)); }

__device__ __forceinline__ v8f wmma_bf(v16bf a, v16bf b, v8f c) {
  return __builtin_amdgcn_wmma_f32_16x16x32_bf16(false, a, false, b, (short)0, c, false, false);
}

// A-operand fragment: 16x32 bf16 tile from row-major [rows, ld], rows at row0..row0+15,
// K slice [k0, k0+32). Lane<16: K chunks [k0,k0+8) & [k0+16,k0+24); lane>=16: +8 each.
__device__ __forceinline__ v16bf load_fragA(const uint16_t* base, int ld, int row0,
                                            int nrows, int k0, int K) {
  int lane = threadIdx.x & 31;
  int r = row0 + (lane & 15);
  int kb = k0 + ((lane >> 4) << 3);
  FragU f;
  f.u[0]=f.u[1]=f.u[2]=f.u[3]=f.u[4]=f.u[5]=f.u[6]=f.u[7]=0u;
  if (r < nrows) {
    const uint16_t* p = base + (size_t)r * (size_t)ld;
    if (kb < K)      { uint4 t = *(const uint4*)(p + kb);      f.u[0]=t.x; f.u[1]=t.y; f.u[2]=t.z; f.u[3]=t.w; }
    if (kb + 16 < K) { uint4 t = *(const uint4*)(p + kb + 16); f.u[4]=t.x; f.u[5]=t.y; f.u[6]=t.z; f.u[7]=t.w; }
  }
  return f.v;
}

// B-operand fragment: B is KxN; we load from Bt (row-major [N, ld>=K]). Lane holds
// column n = lane&15; K chunks [k0,k0+8) & [k0+16,k0+24) for lane<16; lane>=16: +8 each.
__device__ __forceinline__ v16bf load_fragB(const uint16_t* base, int ld, int n0,
                                            int ncols, int k0, int K) {
  int lane = threadIdx.x & 31;
  int n = n0 + (lane & 15);
  int kb = k0 + ((lane >> 4) << 3);
  FragU f;
  f.u[0]=f.u[1]=f.u[2]=f.u[3]=f.u[4]=f.u[5]=f.u[6]=f.u[7]=0u;
  if (n < ncols) {
    const uint16_t* p = base + (size_t)n * (size_t)ld;
    if (kb < K)     { uint4 t = *(const uint4*)(p + kb);     f.u[0]=t.x; f.u[1]=t.y; f.u[2]=t.z; f.u[3]=t.w; }
    if (kb + 16 < K) { uint4 t = *(const uint4*)(p + kb + 16); f.u[4]=t.x; f.u[5]=t.y; f.u[6]=t.z; f.u[7]=t.w; }
  }
  return f.v;
}

// ---------------------------------------------------------------------------
// Generic (optionally batched) WMMA GEMM: C[m,n] (+)= epi( sum_k A[m,k]*Bt[n,k] )
// Block = 256 threads = 8 waves stacked in M; each wave computes an
// (MW*16) x (WN*16) register tile -> MW*WN wmma per 32-deep K step.
// ---------------------------------------------------------------------------
enum { EPI_STORE=0, EPI_SIGMOID=1, EPI_RELU=2, EPI_GATEMUL=3,
       EPI_ACCADD=4, EPI_ADDTO=5, EPI_MULADD=6 };

template<int WN, int MW, int EPI, bool BF16OUT>
__global__ __launch_bounds__(256)
void gemm_bf16(const uint16_t* __restrict__ A,  int lda, long strideA,
               const uint16_t* __restrict__ Bt, int ldb, long strideB,
               void* __restrict__ Cv,           int ldc, long strideC,
               const float* __restrict__ bias,
               const float* __restrict__ other, long strideO,
               int M, int N, int K)
{
  const int batch = blockIdx.z;
  A  += (size_t)batch * (size_t)strideA;
  Bt += (size_t)batch * (size_t)strideB;
  const int wave = threadIdx.x >> 5;
  const int lane = threadIdx.x & 31;
  const int mbase = blockIdx.y * (128 * MW) + wave * (16 * MW);
  const int nbase = blockIdx.x * (16 * WN);

  v8f acc[MW][WN];
#pragma unroll
  for (int mi = 0; mi < MW; ++mi)
#pragma unroll
    for (int j = 0; j < WN; ++j) acc[mi][j] = v8f{0.f,0.f,0.f,0.f,0.f,0.f,0.f,0.f};

  const int KT = (K + 31) >> 5;
  for (int kt = 0; kt < KT; ++kt) {
    const int k0 = kt << 5;
    v16bf a[MW];
#pragma unroll
    for (int mi = 0; mi < MW; ++mi)
      a[mi] = load_fragA(A, lda, mbase + 16*mi, M, k0, K);
    // software prefetch of the next A K-slice (emits global_prefetch)
    if (k0 + 32 < K) {
      const int r = mbase + (lane & 15);
      if (r < M) __builtin_prefetch(A + (size_t)r * (size_t)lda + k0 + 32, 0, 0);
    }
#pragma unroll
    for (int j = 0; j < WN; ++j) {
      v16bf b = load_fragB(Bt, ldb, nbase + 16*j, N, k0, K);
#pragma unroll
      for (int mi = 0; mi < MW; ++mi)
        acc[mi][j] = wmma_bf(a[mi], b, acc[mi][j]);
    }
  }

#pragma unroll
  for (int mi = 0; mi < MW; ++mi) {
    const int mrow = mbase + 16*mi + ((lane >> 4) << 3);
#pragma unroll
    for (int j = 0; j < WN; ++j) {
      const int col = nbase + 16*j + (lane & 15);
      if (col >= N) continue;
      const float bv = bias ? bias[col] : 0.f;
#pragma unroll
      for (int g = 0; g < 8; ++g) {
        const int row = mrow + g;
        if (row >= M) break;
        const size_t idx = (size_t)batch * (size_t)strideC + (size_t)row * (size_t)ldc + col;
        const float v = acc[mi][j][g];
        if (EPI == EPI_STORE) {
          float o = v + bv;
          if (BF16OUT) ((uint16_t*)Cv)[idx] = f2bf(o); else ((float*)Cv)[idx] = o;
        } else if (EPI == EPI_SIGMOID) {
          float o = sigf(v + bv);
          if (BF16OUT) ((uint16_t*)Cv)[idx] = f2bf(o); else ((float*)Cv)[idx] = o;
        } else if (EPI == EPI_RELU) {
          float o = fmaxf(v + bv, 0.f);
          if (BF16OUT) ((uint16_t*)Cv)[idx] = f2bf(o); else ((float*)Cv)[idx] = o;
        } else if (EPI == EPI_GATEMUL) {
          ((float*)Cv)[idx] = sigf(v + bv) * other[idx];
        } else if (EPI == EPI_ACCADD) {
          ((float*)Cv)[idx] += v;
        } else if (EPI == EPI_ADDTO) {
          ((float*)Cv)[idx] += v + bv;
        } else if (EPI == EPI_MULADD) {
          ((float*)Cv)[idx] += other[(size_t)batch*(size_t)strideO + (size_t)row*(size_t)ldc + col] * (v + bv);
        }
      }
    }
  }
}

// ---------------------------------------------------------------------------
// LayerNorm over C=256, f32 in -> bf16 out. One wave per row, 8 rows / block.
// ---------------------------------------------------------------------------
__global__ __launch_bounds__(256)
void ln256_kernel(const float* __restrict__ x, const float* __restrict__ g,
                  const float* __restrict__ b, uint16_t* __restrict__ out, int rows)
{
  const int wave = threadIdx.x >> 5, lane = threadIdx.x & 31;
  const long row = (long)blockIdx.x * 8 + wave;
  if (row >= rows) return;
  const float* xr = x + row * CC;
  float v[8];
#pragma unroll
  for (int t = 0; t < 8; ++t) v[t] = xr[lane * 8 + t];
  float s = 0.f, sq = 0.f;
#pragma unroll
  for (int t = 0; t < 8; ++t) { s += v[t]; sq += v[t]*v[t]; }
#pragma unroll
  for (int m = 1; m < 32; m <<= 1) { s += __shfl_xor(s, m, 32); sq += __shfl_xor(sq, m, 32); }
  const float mean = s * (1.f/CC);
  const float var  = sq * (1.f/CC) - mean*mean;
  const float rs   = rsqrtf(var + 1e-5f);
  uint16_t* orow = out + row * CC + lane*8;
#pragma unroll
  for (int t = 0; t < 8; ++t) {
    int c = lane*8 + t;
    orow[t] = f2bf((v[t] - mean) * rs * g[c] + b[c]);
  }
}

// LN over CT=16 channel planes of blk[c][i*320+j]; writes s_bf[(i*320+j)*16+c].
__global__ __launch_bounds__(256)
void lnc16_kernel(const float* __restrict__ blk, const float* __restrict__ g,
                  const float* __restrict__ b, uint16_t* __restrict__ out)
{
  const long pos = (long)blockIdx.x * 256 + threadIdx.x;
  if (pos >= MROWS) return;
  float v[CT];
  float s = 0.f;
#pragma unroll
  for (int c = 0; c < CT; ++c) { v[c] = blk[(size_t)c * MROWS + pos]; s += v[c]; }
  const float mean = s * (1.f/CT);
  float sq = 0.f;
#pragma unroll
  for (int c = 0; c < CT; ++c) { float d = v[c]-mean; sq += d*d; }
  const float rs = rsqrtf(sq * (1.f/CT) + 1e-5f);
#pragma unroll
  for (int c = 0; c < CT; ++c)
    out[pos*CT + c] = f2bf((v[c]-mean)*rs*g[c] + b[c]);
}

// ---------------------------------------------------------------------------
// Small utility kernels
// ---------------------------------------------------------------------------
__global__ void outer_kernel(const float* __restrict__ n1, const float* __restrict__ n2,
                             float* __restrict__ z)
{
  const long t = (long)blockIdx.x * 256 + threadIdx.x;
  if (t >= (long)MROWS * CC) return;
  const int c = (int)(t & 255);
  const long r = t >> 8;
  const int j = (int)(r % NN2), i = (int)(r / NN2);
  z[t] = n1[i*CC + c] * n2[j*CC + c];
}

__global__ void transpose_bf_kernel(const float* __restrict__ src, uint16_t* __restrict__ dst,
                                    int K, int N)  // src [K,N] -> dst [N,K] bf16
{
  const long t = (long)blockIdx.x * 256 + threadIdx.x;
  if (t >= (long)K * N) return;
  const int n = (int)(t / K), k = (int)(t % K);
  dst[t] = f2bf(src[(long)k * N + n]);
}

__global__ void cvt_bf_kernel(const float* __restrict__ src, uint16_t* __restrict__ dst, long n)
{
  const long t = (long)blockIdx.x * 256 + threadIdx.x;
  if (t < n) dst[t] = f2bf(src[t]);
}

__global__ void scatter_kernel(const float* __restrict__ pp, const int* __restrict__ src,
                               const int* __restrict__ dst, float* __restrict__ d)
{
  const int t = blockIdx.x * 256 + threadIdx.x;
  if (t >= EE * CT) return;
  const int e = t >> 4, c = t & 15;
  atomicAdd(d + (size_t)c * MROWS + (size_t)src[e] * NN2 + dst[e], pp[t]);
}

// abt1[c][j][k] = ab1[(k*320+j)*16+c]  (Bt for block1)
__global__ void perm_abt1_kernel(const float* __restrict__ ab1, uint16_t* __restrict__ abt)
{
  const long t = (long)blockIdx.x * 256 + threadIdx.x;
  if (t >= (long)CT * MROWS) return;
  const int c = (int)(t / MROWS); const long r = t % MROWS;
  const int j = (int)(r / NN1), k = (int)(r % NN1);
  abt[t] = f2bf(ab1[((long)k * NN2 + j) * CT + c]);
}

// ab2cm[c][i][k] = ab2[(i*320+k)*16+c]  (A for block2)
__global__ void perm_ab2_kernel(const float* __restrict__ ab2, uint16_t* __restrict__ out)
{
  const long t = (long)blockIdx.x * 256 + threadIdx.x;
  if (t >= (long)CT * MROWS) return;
  const int c = (int)(t / MROWS); const long r = t % MROWS;
  const int i = (int)(r / NN2), k = (int)(r % NN2);
  out[t] = f2bf(ab2[((long)i * NN2 + k) * CT + c]);
}

// vT[i][h][c][j] = v_bf[(i*320+j)*128 + h*32 + c]
__global__ void packvt_kernel(const uint16_t* __restrict__ v, uint16_t* __restrict__ vT)
{
  const long t = (long)blockIdx.x * 256 + threadIdx.x;
  if (t >= (long)NN1 * HH * CHD * NN2) return;
  const int j = (int)(t % NN2); long r = t / NN2;
  const int c = (int)(r % CHD); r /= CHD;
  const int h = (int)(r % HH);
  const int i = (int)(r / HH);
  vT[t] = v[((long)i * NN2 + j) * (HH*CHD) + h*CHD + c];
}

__global__ void head_kernel(const float* __restrict__ z, const float* __restrict__ w,
                            const float* __restrict__ b, float* __restrict__ out)
{
  const long row = (long)blockIdx.x * 256 + threadIdx.x;
  if (row >= MROWS) return;
  const float* zr = z + row * CC;
  float a0 = 0.f, a1 = 0.f;
  for (int c = 0; c < CC; ++c) { float zv = zr[c]; a0 += zv * w[c*2]; a1 += zv * w[c*2+1]; }
  out[row*2]   = a0 + b[0];
  out[row*2+1] = a1 + b[1];
}

// ---------------------------------------------------------------------------
// Per-(i, head, 16-query-tile) attention: logits (WMMA, K=32) -> softmax -> o = P@V
// One wave per block; logits+probs in LDS.
// ---------------------------------------------------------------------------
__global__ __launch_bounds__(32)
void attn_kernel(const uint16_t* __restrict__ qb, const uint16_t* __restrict__ kb,
                 const uint16_t* __restrict__ vT, const uint16_t* __restrict__ gab,
                 uint16_t* __restrict__ obuf)
{
  const int qt = blockIdx.x, h = blockIdx.y, i = blockIdx.z;
  __shared__ __align__(16) float    lg[16 * 336];
  __shared__ __align__(16) uint16_t pb[16 * 336];
  const int lane = threadIdx.x;
  const int mloc = (lane >> 4) << 3, nn = lane & 15;

  const uint16_t* qi = qb + (size_t)i * NN2 * (HH*CHD) + h * CHD;
  const uint16_t* ki = kb + (size_t)i * NN2 * (HH*CHD) + h * CHD;

  v16bf aq = load_fragA(qi, HH*CHD, qt*16, NN2, 0, CHD);
  for (int kt = 0; kt < NN2/16; ++kt) {
    v16bf bk = load_fragB(ki, HH*CHD, kt*16, NN2, 0, CHD);
    v8f acc = v8f{0.f,0.f,0.f,0.f,0.f,0.f,0.f,0.f};
    acc = wmma_bf(aq, bk, acc);
#pragma unroll
    for (int g = 0; g < 8; ++g) lg[(mloc + g) * 336 + kt*16 + nn] = acc[g];
  }
  __syncthreads();

  const int r = lane & 15, half = lane >> 4;
  const int c0 = half * 160, c1 = c0 + 160;
  float mx = -1e30f;
  for (int c = c0; c < c1; ++c) mx = fmaxf(mx, lg[r*336 + c]);
  mx = fmaxf(mx, __shfl_xor(mx, 16, 32));
  float s = 0.f;
  for (int c = c0; c < c1; ++c) { float e = __expf(lg[r*336 + c] - mx); lg[r*336 + c] = e; s += e; }
  s += __shfl_xor(s, 16, 32);
  const float inv = 1.f / s;
  for (int c = c0; c < c1; ++c) pb[r*336 + c] = f2bf(lg[r*336 + c] * inv);
  __syncthreads();

  const uint16_t* vTi = vT + ((size_t)i * HH + h) * CHD * NN2;
#pragma unroll
  for (int jt = 0; jt < 2; ++jt) {
    v8f acc = v8f{0.f,0.f,0.f,0.f,0.f,0.f,0.f,0.f};
    for (int kt = 0; kt < NN2/32; ++kt) {
      v16bf a = load_fragA(pb, 336, 0, 16, kt*32, NN2);
      v16bf b = load_fragB(vTi, NN2, jt*16, CHD, kt*32, NN2);
      acc = wmma_bf(a, b, acc);
    }
#pragma unroll
    for (int g = 0; g < 8; ++g) {
      const int row = qt*16 + mloc + g;
      const int col = h*CHD + jt*16 + nn;
      const size_t idx = ((size_t)i * NN2 + row) * (HH*CHD) + col;
      obuf[idx] = f2bf(acc[g] * bf2f(gab[idx]));
    }
  }
}

// ---------------------------------------------------------------------------
// Fused Transition: z_out = relu(zl @ w1 + b1) @ w2 + b2, hidden [16,1024] in LDS.
// 16 rows / block, 8 waves; stage1: 64 n-tiles of w1; stage2: 16 n-tiles of w2.
// Input tile staged to LDS via async global->LDS loads when available.
// ---------------------------------------------------------------------------
__global__ __launch_bounds__(256)
void transition_kernel(const uint16_t* __restrict__ zl,
                       const uint16_t* __restrict__ w1T, const float* __restrict__ b1,
                       const uint16_t* __restrict__ w2T, const float* __restrict__ b2,
                       float* __restrict__ z)
{
  __shared__ __align__(16) uint16_t xa[16 * 256];
  __shared__ __align__(16) uint16_t hh[16 * 1024];
  const long rbase = (long)blockIdx.x * 16;

#if HAVE_ASYNC_LDS
  for (int t = threadIdx.x; t < 16*256/8; t += 256) {
    const int r = t >> 5, cu = t & 31;
    __builtin_amdgcn_global_load_async_to_lds_b128(
        (gv4i*)(uintptr_t)(zl + (rbase + r) * CC + cu*8),
        (lv4i*)(&xa[t*8]),
        0, 0);
  }
  __builtin_amdgcn_s_wait_asynccnt(0);
  __syncthreads();
#else
  for (int t = threadIdx.x; t < 16*256/8; t += 256) {
    const int r = t >> 5, cu = t & 31;
    ((uint4*)xa)[t] = *(const uint4*)(zl + (rbase + r) * CC + cu*8);
  }
  __syncthreads();
#endif

  const int wave = threadIdx.x >> 5, lane = threadIdx.x & 31;
  const int mloc = (lane >> 4) << 3, nn = lane & 15;

  for (int nt = wave; nt < 64; nt += 8) {
    v8f acc = v8f{0.f,0.f,0.f,0.f,0.f,0.f,0.f,0.f};
#pragma unroll
    for (int kt = 0; kt < 8; ++kt) {
      v16bf a = load_fragA(xa, 256, 0, 16, kt*32, 256);
      v16bf b = load_fragB(w1T, 256, nt*16, 1024, kt*32, 256);
      acc = wmma_bf(a, b, acc);
    }
    const int col = nt*16 + nn;
    const float bv = b1[col];
#pragma unroll
    for (int g = 0; g < 8; ++g)
      hh[(mloc + g) * 1024 + col] = f2bf(fmaxf(acc[g] + bv, 0.f));
  }
  __syncthreads();

  for (int nt = wave; nt < 16; nt += 8) {
    v8f acc = v8f{0.f,0.f,0.f,0.f,0.f,0.f,0.f,0.f};
    for (int kt = 0; kt < 32; ++kt) {
      v16bf a = load_fragA(hh, 1024, 0, 16, kt*32, 1024);
      v16bf b = load_fragB(w2T, 1024, nt*16, 256, kt*32, 1024);
      acc = wmma_bf(a, b, acc);
    }
    const int col = nt*16 + nn;
    const float bv = b2[col];
#pragma unroll
    for (int g = 0; g < 8; ++g)
      z[(rbase + mloc + g) * CC + col] = acc[g] + bv;
  }
}

// ---------------------------------------------------------------------------
// Host orchestration
// ---------------------------------------------------------------------------
static inline dim3 ggrid(int M, int N, int WN, int MW, int batch) {
  return dim3((N + 16*WN - 1) / (16*WN), (M + 128*MW - 1) / (128*MW), batch);
}
static inline int cdiv(long a, long b) { return (int)((a + b - 1) / b); }

extern "C" void kernel_launch(void* const* d_in, const int* in_sizes, int n_in,
                              void* d_out, int out_size, void* d_ws, size_t ws_size,
                              hipStream_t stream) {
  (void)in_sizes; (void)n_in; (void)out_size; (void)ws_size;
  const float* node1 = (const float*)d_in[0];
  const float* node2 = (const float*)d_in[1];
  const float* ef1   = (const float*)d_in[2];
  const float* ef2   = (const float*)d_in[3];
  const int*   e1s   = (const int*)d_in[4];
  const int*   e1d   = (const int*)d_in[5];
  const int*   e2s   = (const int*)d_in[6];
  const int*   e2d   = (const int*)d_in[7];

  int pi = 8;
  struct AttnP { const float *bf,*bg,*ln_b,*ln_g,*wf,*wg,*wk,*wq,*wv; } attn[2];
  for (int l = 0; l < 2; ++l) {
    attn[l].bf=(const float*)d_in[pi++]; attn[l].bg=(const float*)d_in[pi++];
    attn[l].ln_b=(const float*)d_in[pi++]; attn[l].ln_g=(const float*)d_in[pi++];
    attn[l].wf=(const float*)d_in[pi++]; attn[l].wg=(const float*)d_in[pi++];
    attn[l].wk=(const float*)d_in[pi++]; attn[l].wq=(const float*)d_in[pi++];
    attn[l].wv=(const float*)d_in[pi++];
  }
  const float* out_b = (const float*)d_in[pi++];
  const float* out_w = (const float*)d_in[pi++];
  struct TransP { const float *b1,*b2,*ln_b,*ln_g,*w1,*w2; } trans;
  trans.b1=(const float*)d_in[pi++]; trans.b2=(const float*)d_in[pi++];
  trans.ln_b=(const float*)d_in[pi++]; trans.ln_g=(const float*)d_in[pi++];
  trans.w1=(const float*)d_in[pi++]; trans.w2=(const float*)d_in[pi++];
  struct TrigP { const float *asb,*asw,*b1,*b2,*egb,*egw,*gb1,*gb2,*gw1,*gw2,
                             *ln_b,*ln_g,*lnc_b,*lnc_g,*w1,*w2; } trig[2];
  for (int l = 0; l < 2; ++l) {
    trig[l].asb=(const float*)d_in[pi++]; trig[l].asw=(const float*)d_in[pi++];
    trig[l].b1=(const float*)d_in[pi++]; trig[l].b2=(const float*)d_in[pi++];
    trig[l].egb=(const float*)d_in[pi++]; trig[l].egw=(const float*)d_in[pi++];
    trig[l].gb1=(const float*)d_in[pi++]; trig[l].gb2=(const float*)d_in[pi++];
    trig[l].gw1=(const float*)d_in[pi++]; trig[l].gw2=(const float*)d_in[pi++];
    trig[l].ln_b=(const float*)d_in[pi++]; trig[l].ln_g=(const float*)d_in[pi++];
    trig[l].lnc_b=(const float*)d_in[pi++]; trig[l].lnc_g=(const float*)d_in[pi++];
    trig[l].w1=(const float*)d_in[pi++]; trig[l].w2=(const float*)d_in[pi++];
  }

  // ---- workspace carve-up ----
  char* W = (char*)d_ws;
  size_t off = 0;
  auto take = [&](size_t bytes) -> char* {
    char* p = W + off; off += (bytes + 255) & ~(size_t)255; return p;
  };
  float*    z     = (float*)   take((size_t)MROWS * CC * 4);
  uint16_t* zl    = (uint16_t*)take((size_t)MROWS * CC * 2);
  float*    gbuf  = (float*)   take((size_t)MROWS * CC * 4);
  float*    tmpf  = (float*)   take((size_t)MROWS * CT * 4);
  float*    ab1f  = (float*)   take((size_t)MROWS * CT * 4);
  float*    ab2f  = (float*)   take((size_t)MROWS * CT * 4);
  uint16_t* p1l   = (uint16_t*)take((size_t)EE * CC * 2);
  uint16_t* p2l   = (uint16_t*)take((size_t)EE * CC * 2);
  float*    pp1   = (float*)   take((size_t)EE * CT * 4);
  float*    pp2   = (float*)   take((size_t)EE * CT * 4);
  float*    d1f   = (float*)   take((size_t)CT * MROWS * 4);
  float*    d2f   = (float*)   take((size_t)CT * MROWS * 4);
  uint16_t* d1b   = (uint16_t*)take((size_t)CT * MROWS * 2);
  uint16_t* d2b   = (uint16_t*)take((size_t)CT * MROWS * 2);
  uint16_t* abt1  = (uint16_t*)take((size_t)CT * MROWS * 2);
  uint16_t* ab2cm = (uint16_t*)take((size_t)CT * MROWS * 2);
  float*    blk   = (float*)   take((size_t)CT * MROWS * 4);
  uint16_t* s_bf  = (uint16_t*)take((size_t)MROWS * CT * 2);
  uint16_t* q_bf  = (uint16_t*)take((size_t)MROWS * HH*CHD * 2);
  uint16_t* k_bf  = (uint16_t*)take((size_t)MROWS * HH*CHD * 2);
  uint16_t* v_bf  = (uint16_t*)take((size_t)MROWS * HH*CHD * 2);
  uint16_t* g_bf  = (uint16_t*)take((size_t)MROWS * HH*CHD * 2);
  uint16_t* vT    = (uint16_t*)take((size_t)MROWS * HH*CHD * 2);
  uint16_t* obuf  = (uint16_t*)take((size_t)MROWS * HH*CHD * 2);
  // transposed bf16 weights
  uint16_t *w1T[2], *gw1T[2], *w2T[2], *gw2T[2], *egwT[2], *aswT[2];
  uint16_t *wqT[2], *wkT[2], *wvT[2], *wgT[2], *wfT[2];
  for (int l = 0; l < 2; ++l) {
    w1T[l]  = (uint16_t*)take((size_t)CT*CC*2);  gw1T[l] = (uint16_t*)take((size_t)CT*CC*2);
    w2T[l]  = (uint16_t*)take((size_t)CT*CC*2);  gw2T[l] = (uint16_t*)take((size_t)CT*CC*2);
    egwT[l] = (uint16_t*)take((size_t)CC*CC*2);  aswT[l] = (uint16_t*)take((size_t)CC*CT*2);
    wqT[l]  = (uint16_t*)take((size_t)128*CC*2); wkT[l]  = (uint16_t*)take((size_t)128*CC*2);
    wvT[l]  = (uint16_t*)take((size_t)128*CC*2); wgT[l]  = (uint16_t*)take((size_t)128*CC*2);
    wfT[l]  = (uint16_t*)take((size_t)CC*128*2);
  }
  uint16_t* tw1T = (uint16_t*)take((size_t)1024*CC*2);
  uint16_t* tw2T = (uint16_t*)take((size_t)CC*1024*2);

  // ---- weight prep (transpose + bf16) ----
  auto T = [&](const float* src, uint16_t* dst, int K, int N) {
    transpose_bf_kernel<<<cdiv((long)K*N, 256), 256, 0, stream>>>(src, dst, K, N);
  };
  for (int l = 0; l < 2; ++l) {
    T(trig[l].w1,  w1T[l],  CC, CT);  T(trig[l].gw1, gw1T[l], CC, CT);
    T(trig[l].w2,  w2T[l],  CC, CT);  T(trig[l].gw2, gw2T[l], CC, CT);
    T(trig[l].egw, egwT[l], CC, CC);  T(trig[l].asw, aswT[l], CT, CC);
    T(attn[l].wq, wqT[l], CC, 128);   T(attn[l].wk, wkT[l], CC, 128);
    T(attn[l].wv, wvT[l], CC, 128);   T(attn[l].wg, wgT[l], CC, 128);
    T(attn[l].wf, wfT[l], 128, CC);
  }
  T(trans.w1, tw1T, CC, 1024);
  T(trans.w2, tw2T, 1024, CC);

  // ---- z = outer(node1, node2) ----
  outer_kernel<<<cdiv((long)MROWS*CC, 256), 256, 0, stream>>>(node1, node2, z);

  for (int l = 0; l < 2; ++l) {
    // ================= trig update =================
    ln256_kernel<<<MROWS/8, 256, 0, stream>>>(z,   trig[l].ln_g, trig[l].ln_b, zl,  MROWS);
    ln256_kernel<<<EE/8,    256, 0, stream>>>(ef1, trig[l].ln_g, trig[l].ln_b, p1l, EE);
    ln256_kernel<<<EE/8,    256, 0, stream>>>(ef2, trig[l].ln_g, trig[l].ln_b, p2l, EE);

    // ab1 = sigmoid(zl@gw1+gb1) * (zl@w1+b1) ; ab2 similarly
    gemm_bf16<1,2,EPI_STORE,false><<<ggrid(MROWS,CT,1,2,1),256,0,stream>>>(
        zl,CC,0, w1T[l],CC,0, tmpf,CT,0, trig[l].b1, nullptr,0, MROWS,CT,CC);
    gemm_bf16<1,2,EPI_GATEMUL,false><<<ggrid(MROWS,CT,1,2,1),256,0,stream>>>(
        zl,CC,0, gw1T[l],CC,0, ab1f,CT,0, trig[l].gb1, tmpf,0, MROWS,CT,CC);
    gemm_bf16<1,2,EPI_STORE,false><<<ggrid(MROWS,CT,1,2,1),256,0,stream>>>(
        zl,CC,0, w2T[l],CC,0, tmpf,CT,0, trig[l].b2, nullptr,0, MROWS,CT,CC);
    gemm_bf16<1,2,EPI_GATEMUL,false><<<ggrid(MROWS,CT,1,2,1),256,0,stream>>>(
        zl,CC,0, gw2T[l],CC,0, ab2f,CT,0, trig[l].gb2, tmpf,0, MROWS,CT,CC);

    // pp1/pp2 on edge features (same weights)
    gemm_bf16<1,2,EPI_STORE,false><<<ggrid(EE,CT,1,2,1),256,0,stream>>>(
        p1l,CC,0, w1T[l],CC,0, tmpf,CT,0, trig[l].b1, nullptr,0, EE,CT,CC);
    gemm_bf16<1,2,EPI_GATEMUL,false><<<ggrid(EE,CT,1,2,1),256,0,stream>>>(
        p1l,CC,0, gw1T[l],CC,0, pp1,CT,0, trig[l].gb1, tmpf,0, EE,CT,CC);
    gemm_bf16<1,2,EPI_STORE,false><<<ggrid(EE,CT,1,2,1),256,0,stream>>>(
        p2l,CC,0, w2T[l],CC,0, tmpf,CT,0, trig[l].b2, nullptr,0, EE,CT,CC);
    gemm_bf16<1,2,EPI_GATEMUL,false><<<ggrid(EE,CT,1,2,1),256,0,stream>>>(
        p2l,CC,0, gw2T[l],CC,0, pp2,CT,0, trig[l].gb2, tmpf,0, EE,CT,CC);

    // g = sigmoid(zl@egw + egb)
    gemm_bf16<4,2,EPI_SIGMOID,false><<<ggrid(MROWS,CC,4,2,1),256,0,stream>>>(
        zl,CC,0, egwT[l],CC,0, gbuf,CC,0, trig[l].egb, nullptr,0, MROWS,CC,CC);

    // dense scatter of edge tensors
    (void)hipMemsetAsync(d1f, 0, (size_t)CT*MROWS*4, stream);
    (void)hipMemsetAsync(d2f, 0, (size_t)CT*MROWS*4, stream);
    scatter_kernel<<<cdiv((long)EE*CT,256),256,0,stream>>>(pp1, e1s, e1d, d1f);
    scatter_kernel<<<cdiv((long)EE*CT,256),256,0,stream>>>(pp2, e2s, e2d, d2f);
    cvt_bf_kernel<<<cdiv((long)CT*MROWS,256),256,0,stream>>>(d1f, d1b, (long)CT*MROWS);
    cvt_bf_kernel<<<cdiv((long)CT*MROWS,256),256,0,stream>>>(d2f, d2b, (long)CT*MROWS);
    perm_abt1_kernel<<<cdiv((long)CT*MROWS,256),256,0,stream>>>(ab1f, abt1);
    perm_ab2_kernel <<<cdiv((long)CT*MROWS,256),256,0,stream>>>(ab2f, ab2cm);

    // blk[c] = d1[c] @ ab1[c]  ;  blk[c] += ab2[c] @ d2[c]^T   (batched over 16 channels)
    gemm_bf16<4,2,EPI_STORE,false><<<ggrid(NN1,NN2,4,2,CT),256,0,stream>>>(
        d1b,NN1,(long)MROWS, abt1,NN1,(long)MROWS, blk,NN2,(long)MROWS,
        nullptr, nullptr,0, NN1,NN2,NN1);
    gemm_bf16<4,2,EPI_ACCADD,false><<<ggrid(NN1,NN2,4,2,CT),256,0,stream>>>(
        ab2cm,NN2,(long)MROWS, d2b,NN2,(long)MROWS, blk,NN2,(long)MROWS,
        nullptr, nullptr,0, NN1,NN2,NN2);

    // s = LN_ct(blk) ;  z += g * (s @ asw + asb)
    lnc16_kernel<<<cdiv(MROWS,256),256,0,stream>>>(blk, trig[l].lnc_g, trig[l].lnc_b, s_bf);
    gemm_bf16<4,2,EPI_MULADD,false><<<ggrid(MROWS,CC,4,2,1),256,0,stream>>>(
        s_bf,CT,0, aswT[l],CT,0, z,CC,0, trig[l].asb, gbuf,0, MROWS,CC,CT);

    // ================= triangle attention =================
    ln256_kernel<<<MROWS/8, 256, 0, stream>>>(z, attn[l].ln_g, attn[l].ln_b, zl, MROWS);
    gemm_bf16<4,2,EPI_STORE,true><<<ggrid(MROWS,128,4,2,1),256,0,stream>>>(
        zl,CC,0, wqT[l],CC,0, q_bf,128,0, nullptr, nullptr,0, MROWS,128,CC);
    gemm_bf16<4,2,EPI_STORE,true><<<ggrid(MROWS,128,4,2,1),256,0,stream>>>(
        zl,CC,0, wkT[l],CC,0, k_bf,128,0, nullptr, nullptr,0, MROWS,128,CC);
    gemm_bf16<4,2,EPI_STORE,true><<<ggrid(MROWS,128,4,2,1),256,0,stream>>>(
        zl,CC,0, wvT[l],CC,0, v_bf,128,0, nullptr, nullptr,0, MROWS,128,CC);
    gemm_bf16<4,2,EPI_SIGMOID,true><<<ggrid(MROWS,128,4,2,1),256,0,stream>>>(
        zl,CC,0, wgT[l],CC,0, g_bf,128,0, attn[l].bg, nullptr,0, MROWS,128,CC);
    packvt_kernel<<<cdiv((long)MROWS*128,256),256,0,stream>>>(v_bf, vT);
    attn_kernel<<<dim3(NN2/16, HH, NN1), 32, 0, stream>>>(q_bf, k_bf, vT, g_bf, obuf);
    gemm_bf16<4,2,EPI_ADDTO,false><<<ggrid(MROWS,CC,4,2,1),256,0,stream>>>(
        obuf,128,0, wfT[l],128,0, z,CC,0, attn[l].bf, nullptr,0, MROWS,CC,128);

    // ================= transition (shared params) =================
    ln256_kernel<<<MROWS/8, 256, 0, stream>>>(z, trans.ln_g, trans.ln_b, zl, MROWS);
    transition_kernel<<<MROWS/16, 256, 0, stream>>>(zl, tw1T, trans.b1, tw2T, trans.b2, z);
  }

  // ---- output head [M,2] ----
  head_kernel<<<cdiv(MROWS,256),256,0,stream>>>(z, out_w, out_b, (float*)d_out);
}
